// AllAtomEquivariantAtomConvolution_8461085573573
// MI455X (gfx1250) — matrix operand order repaired and checked by reading, not compiled, not run on hardware
//
#include <hip/hip_runtime.h>
#include <math.h>

// ---------------- problem constants ----------------
#define FLAT 2048          // B*N
#define EQ 131072
#define NEQ 64
#define ATOM 128
#define NODE0 32
#define NODE1 16
#define OUT0 16
#define OUT1 8
#define EDIM 32
#define HID 128
#define LAT 128
#define RBF 16
#define ZEMB 32
#define WNUM 1152          // 512 + 256 + 256 + 128
#define IRR 40             // OUT0 + 3*OUT1
#define CUTOFF 5.0f
#define PI_F 3.14159265358979f
#define SQRT3_F 1.73205080757f
#define INV_FAN 0.14433756729740643f   // 1/sqrt(48)

// ---------------- vector / fragment types ----------------
typedef __attribute__((ext_vector_type(16))) __bf16 v16bf;
typedef __attribute__((ext_vector_type(8)))  float  v8f;
typedef __attribute__((ext_vector_type(4)))  unsigned int u32x4;
typedef __attribute__((ext_vector_type(4)))  float  f32x4;

union Frag16 { v16bf v; __bf16 e[16]; u32x4 q[2]; };
union CFrag  { v8f   v; float  e[8];  };

__device__ __forceinline__ u32x4 ldq(const __bf16* p) { return *(const u32x4*)p; }
__device__ __forceinline__ f32x4 ldf4(const float* p) { return *(const f32x4*)p; }
__device__ __forceinline__ int jmapA(int j) { return (j < 8) ? j : j + 8; }
__device__ __forceinline__ float siluf(float x) {
    return x * __builtin_amdgcn_rcpf(1.f + __expf(-x));
}
__device__ __forceinline__ float sigmf(float x) {
    return __builtin_amdgcn_rcpf(1.f + __expf(-x));
}

#define WMMA_BF16(A, B, C) \
    __builtin_amdgcn_wmma_f32_16x16x32_bf16(false, (A), false, (B), (short)0, (C), false, false)

// Load an A fragment (16x32 tile row) from a bf16 row pointer at K offset k0.
// ISA layout: j=0..7 -> K = akb+k0 .. +7 ; j=8..15 -> K = akb+k0+16 .. +23
__device__ __forceinline__ Frag16 loadA(const __bf16* row, int k0, int akb) {
    Frag16 a;
    a.q[0] = ldq(row + k0 + akb);
    a.q[1] = ldq(row + k0 + akb + 16);
    return a;
}
// Load a B fragment from LDS weights staged transposed as [col][k] (stride kp).
// ISA layout: j=0..7 -> K = bkb/2+k0 .. +7 ; j=8..15 -> K = bkb/2+k0+16 .. +23
__device__ __forceinline__ Frag16 loadB(const __bf16* wt, int col, int kp, int k0, int bkb) {
    Frag16 b;
    const __bf16* p = wt + col * kp + k0 + (bkb >> 1);
    b.q[0] = ldq(p);
    b.q[1] = ldq(p + 16);
    return b;
}

// ---------------- K0: zero fill ----------------
__global__ void zero_kernel(float* __restrict__ p, int n) {
    int i = blockIdx.x * 256 + threadIdx.x;
    if (i < n) p[i] = 0.f;
}

// ---------------- K1: per-edge prep ----------------
__global__ __launch_bounds__(256) void edge_prep_kernel(
    const float* __restrict__ att_dist, const float* __restrict__ att_vec,
    const int* __restrict__ z, const unsigned char* __restrict__ mask,
    const int* __restrict__ att_src, const int* __restrict__ att_dst,
    const float* __restrict__ z_emb_W,
    float* __restrict__ rbf_buf, float* __restrict__ sh1_buf,
    float* __restrict__ envact, __bf16* __restrict__ win)
{
    int e = blockIdx.x * 256 + threadIdx.x;
    if (e >= EQ) return;
    int src = att_src[e], dst = att_dst[e];
    float dist = att_dist[e];
    bool self = (src == dst);
    float d = fmaxf(dist, 1e-8f);
    float rd = __builtin_amdgcn_rcpf(d);
    float ux, uy, uz;
    if (self) { ux = 0.f; uy = 0.f; uz = 1.f; }
    else { ux = att_vec[e*3+0]*rd; uy = att_vec[e*3+1]*rd; uz = att_vec[e*3+2]*rd; }
    float sh0 = self ? 0.f : SQRT3_F * uy;
    float sh1v = self ? 0.f : SQRT3_F * uz;
    float sh2 = self ? 0.f : SQRT3_F * ux;
    sh1_buf[e*3+0] = sh0; sh1_buf[e*3+1] = sh1v; sh1_buf[e*3+2] = sh2;

    const float width = CUTOFF / 15.f;
    float rb[RBF];
    #pragma unroll
    for (int i = 0; i < RBF; ++i) {
        float c = (float)i * width;
        float t = (dist - c) / width;
        rb[i] = __expf(-0.5f * t * t);
        rbf_buf[e*RBF + i] = rb[i];
    }
    float env = (dist < CUTOFF) ? 0.5f * (cosf(PI_F * dist / CUTOFF) + 1.f) : 0.f;
    float act = (mask[src] && mask[dst]) ? 1.f : 0.f;
    envact[e] = env * act;

    // weight_in = [z_emb(32), is_self(1), rbf(16), pad(15)] as bf16, K padded to 64
    const float* ze = z_emb_W + (long)z[dst] * ZEMB;
    __bf16* wr = win + (long)e * 64;
    #pragma unroll
    for (int k = 0; k < ZEMB; ++k) wr[k] = (__bf16)ze[k];
    wr[32] = (__bf16)(self ? 1.f : 0.f);
    #pragma unroll
    for (int i = 0; i < RBF; ++i) wr[33 + i] = (__bf16)rb[i];
    #pragma unroll
    for (int k = 49; k < 64; ++k) wr[k] = (__bf16)0.f;
}

// ---------------- K2: hidden = silu(win @ vw_W1 + b1), WMMA ----------------
__global__ __launch_bounds__(128) void vw_hidden_kernel(
    const __bf16* __restrict__ win, const float* __restrict__ vw_W1,
    const float* __restrict__ vw_b1, __bf16* __restrict__ hidden1)
{
    __shared__ __align__(16) __bf16 sW1t[HID * 64];   // [col][k], k padded to 64
    __shared__ float sb1[HID];
    int tid = threadIdx.x;
    for (int idx = tid; idx < 64 * HID; idx += 128) {
        int k = idx / HID, c = idx % HID;
        sW1t[c * 64 + k] = (__bf16)((k < 49) ? vw_W1[k*HID + c] : 0.f);
    }
    if (tid < HID) sb1[tid] = vw_b1[tid];
    __syncthreads();

    int wave = tid >> 5, lane = tid & 31;
    int ebase = blockIdx.x * 64 + wave * 16;
    int arow = lane & 15;
    int akb = (lane < 16) ? 0 : 8;
    int col0 = lane & 15;
    int bkb = (lane < 16) ? 0 : 16;
    int rowbase = (lane < 16) ? 0 : 8;

    const __bf16* wrow = win + (long)(ebase + arow) * 64;
    Frag16 a0 = loadA(wrow, 0, akb);
    Frag16 a1 = loadA(wrow, 32, akb);

    for (int ct = 0; ct < 8; ++ct) {
        int col = ct * 16 + col0;
        Frag16 b0 = loadB(sW1t, col, 64, 0, bkb);
        Frag16 b1f = loadB(sW1t, col, 64, 32, bkb);
        CFrag c;
        #pragma unroll
        for (int r = 0; r < 8; ++r) c.e[r] = 0.f;
        c.v = WMMA_BF16(a0.v, b0.v, c.v);
        c.v = WMMA_BF16(a1.v, b1f.v, c.v);
        float bias = sb1[col];
        #pragma unroll
        for (int r = 0; r < 8; ++r) {
            float x = c.e[r] + bias;
            hidden1[(long)(ebase + rowbase + r) * HID + col] = (__bf16)siluf(x);
        }
    }
}

// ---------------- K3: gate MLP, WMMA (K padded 273->288) ----------------
__global__ __launch_bounds__(128) void gate_kernel(
    const float* __restrict__ h, const float* __restrict__ gate_W1,
    const float* __restrict__ gate_b1, const float* __restrict__ gate_W2,
    const float* __restrict__ gate_b2, const float* __restrict__ rbf_buf,
    const float* __restrict__ envact, const int* __restrict__ att_src,
    const int* __restrict__ att_dst, float* __restrict__ we_buf)
{
    __shared__ __align__(16) __bf16 sGt[HID * 32];    // per-K-step slice, [col][kk]
    __shared__ float gpart[4][32][8];
    int tid = threadIdx.x;
    int wave = tid >> 5, lane = tid & 31;
    int ebase = blockIdx.x * 64 + wave * 16;
    int arow = lane & 15;
    int ea = ebase + arow;
    int src = att_src[ea], dst = att_dst[ea];
    float selff = (src == dst) ? 1.f : 0.f;
    const float* hs = h + (long)src * ATOM;
    const float* hd = h + (long)dst * ATOM;
    const float* rb = rbf_buf + (long)ea * RBF;
    int akb = (lane < 16) ? 0 : 8;
    int col0 = lane & 15;
    int bkb = (lane < 16) ? 0 : 16;

    CFrag acc[8];
    #pragma unroll
    for (int ct = 0; ct < 8; ++ct)
        #pragma unroll
        for (int r = 0; r < 8; ++r) acc[ct].e[r] = 0.f;

    for (int ks = 0; ks < 9; ++ks) {
        // stage 32x128 K-slice of gate_W1, transposed, bf16
        for (int idx = tid; idx < 32 * HID; idx += 128) {
            int kk = idx / HID, c = idx % HID;
            int k = ks * 32 + kk;
            sGt[c * 32 + kk] = (__bf16)((k < 273) ? gate_W1[k * HID + c] : 0.f);
        }
        __syncthreads();

        Frag16 a;
        if (ks < 8) {   // contiguous f32 source: h_src (ks<4) or h_dst
            const float* base = (ks < 4) ? (hs + ks * 32) : (hd + (ks - 4) * 32);
            f32x4 p0 = ldf4(base + akb);
            f32x4 p1 = ldf4(base + akb + 4);
            f32x4 p2 = ldf4(base + akb + 16);
            f32x4 p3 = ldf4(base + akb + 20);
            #pragma unroll
            for (int i = 0; i < 4; ++i) {
                a.e[i]      = (__bf16)p0[i];
                a.e[4 + i]  = (__bf16)p1[i];
                a.e[8 + i]  = (__bf16)p2[i];
                a.e[12 + i] = (__bf16)p3[i];
            }
        } else {        // tail: rbf(16) + is_self + zero pad
            #pragma unroll
            for (int j = 0; j < 16; ++j) {
                int k = 256 + akb + jmapA(j);
                float v;
                if (k < 272)       v = rb[k - 256];
                else if (k == 272) v = selff;
                else               v = 0.f;
                a.e[j] = (__bf16)v;
            }
        }
        #pragma unroll
        for (int ct = 0; ct < 8; ++ct) {
            Frag16 b = loadB(sGt, ct * 16 + col0, 32, 0, bkb);
            acc[ct].v = WMMA_BF16(a.v, b.v, acc[ct].v);
        }
        __syncthreads();
    }
    float p[8];
    #pragma unroll
    for (int r = 0; r < 8; ++r) p[r] = 0.f;
    for (int ct = 0; ct < 8; ++ct) {
        int col = ct * 16 + col0;
        float gb = gate_b1[col];
        float gw = gate_W2[col];
        #pragma unroll
        for (int r = 0; r < 8; ++r) p[r] += siluf(acc[ct].e[r] + gb) * gw;
    }
    #pragma unroll
    for (int r = 0; r < 8; ++r) gpart[wave][lane][r] = p[r];
    __syncthreads();
    if (lane < 16) {
        int R = lane;
        int lo = (R < 8) ? 0 : 16;
        int rr = R & 7;
        float s = 0.f;
        for (int l = 0; l < 16; ++l) s += gpart[wave][lo + l][rr];
        float g = sigmf(s + gate_b2[0]);
        int ee = ebase + R;
        we_buf[ee] = envact[ee] * g;
    }
}

// ---------------- K5: edge-feature scales (tiny, VALU) ----------------
__global__ void scales_kernel(
    const float* __restrict__ e_feat, const float* __restrict__ em_W1,
    const float* __restrict__ em_b1, const float* __restrict__ em_W2,
    const float* __restrict__ em_b2, float* __restrict__ scale_full)
{
    int ne = threadIdx.x;
    if (ne >= NEQ) return;
    float o[24];
    for (int j = 0; j < 24; ++j) o[j] = em_b2[j];
    const float* ef = e_feat + ne * EDIM;
    for (int hh = 0; hh < HID; ++hh) {
        float s = em_b1[hh];
        for (int k = 0; k < EDIM; ++k) s += ef[k] * em_W1[k * HID + hh];
        s = siluf(s);
        for (int j = 0; j < 24; ++j) o[j] += s * em_W2[hh * 24 + j];
    }
    float* sf = scale_full + ne * IRR;
    for (int j = 0; j < OUT0; ++j) sf[j] = o[j];
    for (int v = 0; v < OUT1; ++v)
        for (int i = 0; i < 3; ++i) sf[OUT0 + v * 3 + i] = o[OUT0 + v];
}

// ---------------- K4: tw GEMM (E x 128 x 1152) fused with equivariant
//                  contraction + gated segment-sum scatter ----------------
__global__ __launch_bounds__(128) void edge_contract_kernel(
    const __bf16* __restrict__ hidden1, const float* __restrict__ vw_W2,
    const float* __restrict__ vw_b2, const float* __restrict__ h_full,
    const float* __restrict__ sh1_buf, const float* __restrict__ we_buf,
    const int* __restrict__ att_src, const int* __restrict__ att_dst,
    float* __restrict__ out_irrep)
{
    __shared__ __align__(16) __bf16 sW2t[16 * HID];  // tile transposed: [c][k]
    __shared__ float  sb2[16];
    __shared__ float  s_s[4][16][NODE0];
    __shared__ float  s_vd[4][16][NODE1];
    __shared__ float  s_vin[4][16][NODE1][3];
    __shared__ float  s_sh1[4][16][3];
    __shared__ float  s_we[4][16];
    __shared__ int    s_src[4][16];
    __shared__ float  s_o1[4][16][OUT1][3];

    int tid = threadIdx.x;
    int wave = tid >> 5, lane = tid & 31;
    int ebase = blockIdx.x * 64 + wave * 16;
    int e_loc = lane & 15;
    int e = ebase + e_loc;
    int dst = att_dst[e];
    if (lane < 16) {
        s_src[wave][e_loc] = att_src[e];
        s_we[wave][e_loc] = we_buf[e];
        for (int i = 0; i < 3; ++i) s_sh1[wave][e_loc][i] = sh1_buf[e*3 + i];
        const float* hf = h_full + (long)dst * (NODE0 + 3*NODE1);
        for (int u = 0; u < NODE0; ++u) s_s[wave][e_loc][u] = hf[u];
    } else {
        const float* hf = h_full + (long)dst * (NODE0 + 3*NODE1) + NODE0;
        for (int u = 0; u < NODE1; ++u)
            for (int i = 0; i < 3; ++i) s_vin[wave][e_loc][u][i] = hf[u*3 + i];
    }
    for (int idx = tid; idx < 4 * 16 * OUT1 * 3; idx += 128)
        ((float*)s_o1)[idx] = 0.f;
    __syncthreads();
    if (lane >= 16) {
        const float* sh = s_sh1[wave][e_loc];
        for (int u = 0; u < NODE1; ++u) {
            float dv = s_vin[wave][e_loc][u][0]*sh[0]
                     + s_vin[wave][e_loc][u][1]*sh[1]
                     + s_vin[wave][e_loc][u][2]*sh[2];
            s_vd[wave][e_loc][u] = dv * (1.f / SQRT3_F);
        }
    }
    __syncthreads();

    int arow = lane & 15;
    int akb = (lane < 16) ? 0 : 8;
    int col0 = lane & 15;
    int bkb = (lane < 16) ? 0 : 16;
    int rowbase = (lane < 16) ? 0 : 8;

    // A fragments (hidden row, 4 K-steps of 32) — vectorized b128 loads
    const __bf16* hrow = hidden1 + (long)(ebase + arow) * HID;
    Frag16 af[4];
    #pragma unroll
    for (int ks = 0; ks < 4; ++ks) af[ks] = loadA(hrow, ks * 32, akb);

    float o0[8];
    float ao1[8][3];
    #pragma unroll
    for (int r = 0; r < 8; ++r) {
        o0[r] = 0.f;
        ao1[r][0] = 0.f; ao1[r][1] = 0.f; ao1[r][2] = 0.f;
    }

    for (int ct = 0; ct < 72; ++ct) {
        // stage W2 column tile (128 x 16) transposed as bf16
        for (int idx = tid; idx < HID * 16; idx += 128) {
            int k = idx >> 4, c = idx & 15;
            sW2t[c * HID + k] = (__bf16)vw_W2[k * WNUM + ct * 16 + c];
        }
        if (tid < 16) sb2[tid] = vw_b2[ct * 16 + tid];
        if (ct + 1 < 72)  // prefetch next tile (global_prefetch_b8)
            __builtin_prefetch(&vw_W2[tid * WNUM + (ct + 1) * 16], 0, 1);
        __syncthreads();

        CFrag acc;
        #pragma unroll
        for (int r = 0; r < 8; ++r) acc.e[r] = 0.f;
        #pragma unroll
        for (int ks = 0; ks < 4; ++ks) {
            Frag16 b = loadB(sW2t, col0, HID, ks * 32, bkb);
            acc.v = WMMA_BF16(af[ks].v, b.v, acc.v);
        }
        float bias = sb2[col0];
        if (ct < 32) {                       // w000: out0[v] += s_in[u]*tw
            int u = ct;
            #pragma unroll
            for (int r = 0; r < 8; ++r)
                o0[r] += s_s[wave][rowbase + r][u] * (acc.e[r] + bias);
        } else if (ct < 48) {                // w110: out0[v] += (v.sh1)/sqrt3 * tw
            int u = ct - 32;
            #pragma unroll
            for (int r = 0; r < 8; ++r)
                o0[r] += s_vd[wave][rowbase + r][u] * (acc.e[r] + bias);
        } else if (ct < 64) {                // w011: out1[v,i] += s_in[u]*sh1[i]*tw
            int u = (ct - 48) * 2 + (col0 >> 3);
            #pragma unroll
            for (int r = 0; r < 8; ++r) {
                int row = rowbase + r;
                float cf = s_s[wave][row][u] * (acc.e[r] + bias);
                ao1[r][0] += cf * s_sh1[wave][row][0];
                ao1[r][1] += cf * s_sh1[wave][row][1];
                ao1[r][2] += cf * s_sh1[wave][row][2];
            }
        } else {                             // w101: out1[v,i] += v_in[u,i]*tw
            int u = (ct - 64) * 2 + (col0 >> 3);
            #pragma unroll
            for (int r = 0; r < 8; ++r) {
                int row = rowbase + r;
                float t = acc.e[r] + bias;
                ao1[r][0] += s_vin[wave][row][u][0] * t;
                ao1[r][1] += s_vin[wave][row][u][1] * t;
                ao1[r][2] += s_vin[wave][row][u][2] * t;
            }
        }
        __syncthreads();
    }

    // scatter out0 (unique (edge, v) per lane)
    #pragma unroll
    for (int r = 0; r < 8; ++r) {
        int row = rowbase + r;
        float val = o0[r] * INV_FAN * s_we[wave][row];
        atomicAdd(&out_irrep[(long)s_src[wave][row] * IRR + col0], val);
    }
    // reduce out1 pairs (lane, lane^8) via LDS atomics, then scatter
    int v = col0 & 7;
    #pragma unroll
    for (int r = 0; r < 8; ++r) {
        int row = rowbase + r;
        atomicAdd(&s_o1[wave][row][v][0], ao1[r][0]);
        atomicAdd(&s_o1[wave][row][v][1], ao1[r][1]);
        atomicAdd(&s_o1[wave][row][v][2], ao1[r][2]);
    }
    __syncthreads();
    for (int idx = lane; idx < 16 * OUT1 * 3; idx += 32) {
        int row = idx / 24;
        int jj = idx % 24;
        int vv = jj / 3, ii = jj % 3;
        float val = s_o1[wave][row][vv][ii] * INV_FAN * s_we[wave][row];
        atomicAdd(&out_irrep[(long)s_src[wave][row] * IRR + OUT0 + vv*3 + ii], val);
    }
}

// ---------------- K6: output MLP 24->128->128->128, 3 WMMA layers ----------------
__global__ __launch_bounds__(128) void out_mlp_kernel(
    const float* __restrict__ out_irrep, const float* __restrict__ scale_full,
    const float* __restrict__ W1, const float* __restrict__ b1,
    const float* __restrict__ W2, const float* __restrict__ b2,
    const float* __restrict__ W3, const float* __restrict__ b3,
    float* __restrict__ out)
{
    __shared__ __align__(16) __bf16 sWt[HID * HID];  // weights transposed [c][k]
    __shared__ float  sb[HID];
    __shared__ __align__(16) __bf16 sIv[4][16][32];  // invariants, bf16, padded
    __shared__ __align__(16) __bf16 sX[4][16 * HID];

    int tid = threadIdx.x;
    int wave = tid >> 5, lane = tid & 31;
    int gbase = blockIdx.x * 64 + wave * 16;

    // build invariants (24, padded to 32) in bf16
    for (int idx = lane; idx < 16 * 24; idx += 32) {
        int rl = idx / 24, j = idx % 24;
        int grow = gbase + rl;
        int node = grow >> 6;      // / NE
        int ne = grow & 63;
        const float* ir = out_irrep + (long)node * IRR;
        const float* sf = scale_full + ne * IRR;
        float v;
        if (j < OUT0) v = ir[j] * sf[j];
        else {
            int vv = j - OUT0;
            float a = 1e-12f;
            for (int i = 0; i < 3; ++i) {
                float t = ir[OUT0 + vv*3 + i] * sf[OUT0 + vv*3 + i];
                a += t * t;
            }
            v = sqrtf(a);
        }
        sIv[wave][rl][j] = (__bf16)v;
    }
    for (int idx = lane; idx < 16 * 8; idx += 32)
        sIv[wave][idx / 8][24 + (idx & 7)] = (__bf16)0.f;

    // stage W1 (24x128 padded to 32x128), transposed [c][k] with k-stride HID
    for (int idx = tid; idx < 32 * HID; idx += 128) {
        int k = idx / HID, c = idx % HID;
        sWt[c * HID + k] = (__bf16)((k < 24) ? W1[k * HID + c] : 0.f);
    }
    if (tid < HID) sb[tid] = b1[tid];
    __syncthreads();

    int arow = lane & 15;
    int akb = (lane < 16) ? 0 : 8;
    int col0 = lane & 15;
    int bkb = (lane < 16) ? 0 : 16;
    int rowbase = (lane < 16) ? 0 : 8;

    // layer 1 (K=32)
    {
        Frag16 a = loadA(&sIv[wave][arow][0], 0, akb);
        for (int ct = 0; ct < 8; ++ct) {
            int col = ct * 16 + col0;
            Frag16 b = loadB(sWt, col, HID, 0, bkb);
            CFrag c;
            #pragma unroll
            for (int r = 0; r < 8; ++r) c.e[r] = 0.f;
            c.v = WMMA_BF16(a.v, b.v, c.v);
            float bias = sb[col];
            #pragma unroll
            for (int r = 0; r < 8; ++r)
                sX[wave][(rowbase + r) * HID + col] = (__bf16)siluf(c.e[r] + bias);
        }
    }
    __syncthreads();
    // stage W2 (128x128) transposed
    for (int idx = tid; idx < HID * HID; idx += 128) {
        int k = idx / HID, c = idx % HID;
        sWt[c * HID + k] = (__bf16)W2[idx];
    }
    if (tid < HID) sb[tid] = b2[tid];
    __syncthreads();
    // layer 2 (K=128)
    {
        Frag16 af[4];
        #pragma unroll
        for (int ks = 0; ks < 4; ++ks)
            af[ks] = loadA(&sX[wave][arow * HID], ks * 32, akb);
        for (int ct = 0; ct < 8; ++ct) {
            int col = ct * 16 + col0;
            CFrag c;
            #pragma unroll
            for (int r = 0; r < 8; ++r) c.e[r] = 0.f;
            #pragma unroll
            for (int ks = 0; ks < 4; ++ks) {
                Frag16 b = loadB(sWt, col, HID, ks * 32, bkb);
                c.v = WMMA_BF16(af[ks].v, b.v, c.v);
            }
            float bias = sb[col];
            #pragma unroll
            for (int r = 0; r < 8; ++r)
                sX[wave][(rowbase + r) * HID + col] = (__bf16)siluf(c.e[r] + bias);
        }
    }
    __syncthreads();
    // stage W3 transposed
    for (int idx = tid; idx < HID * HID; idx += 128) {
        int k = idx / HID, c = idx % HID;
        sWt[c * HID + k] = (__bf16)W3[idx];
    }
    if (tid < HID) sb[tid] = b3[tid];
    __syncthreads();
    // layer 3 (K=128) -> global out (f32)
    {
        Frag16 af[4];
        #pragma unroll
        for (int ks = 0; ks < 4; ++ks)
            af[ks] = loadA(&sX[wave][arow * HID], ks * 32, akb);
        for (int ct = 0; ct < 8; ++ct) {
            int col = ct * 16 + col0;
            CFrag c;
            #pragma unroll
            for (int r = 0; r < 8; ++r) c.e[r] = 0.f;
            #pragma unroll
            for (int ks = 0; ks < 4; ++ks) {
                Frag16 b = loadB(sWt, col, HID, ks * 32, bkb);
                c.v = WMMA_BF16(af[ks].v, b.v, c.v);
            }
            float bias = sb[col];
            #pragma unroll
            for (int r = 0; r < 8; ++r) {
                int grow = gbase + rowbase + r;
                out[(long)grow * LAT + col] = c.e[r] + bias;
            }
        }
    }
}

// ---------------- host launch ----------------
extern "C" void kernel_launch(void* const* d_in, const int* in_sizes, int n_in,
                              void* d_out, int out_size, void* d_ws, size_t ws_size,
                              hipStream_t stream) {
    const float* h        = (const float*)d_in[0];
    const float* h_full   = (const float*)d_in[1];
    const float* e_feat   = (const float*)d_in[2];
    const float* att_dist = (const float*)d_in[3];
    const float* att_vec  = (const float*)d_in[4];
    const int*   z        = (const int*)d_in[5];
    const unsigned char* mask = (const unsigned char*)d_in[6];
    const int*   att_src  = (const int*)d_in[7];
    const int*   att_dst  = (const int*)d_in[8];
    const float* z_emb_W  = (const float*)d_in[9];
    const float* vw_W1    = (const float*)d_in[10];
    const float* vw_b1    = (const float*)d_in[11];
    const float* vw_W2    = (const float*)d_in[12];
    const float* vw_b2    = (const float*)d_in[13];
    const float* gate_W1  = (const float*)d_in[14];
    const float* gate_b1  = (const float*)d_in[15];
    const float* gate_W2  = (const float*)d_in[16];
    const float* gate_b2  = (const float*)d_in[17];
    const float* em_W1    = (const float*)d_in[18];
    const float* em_b1    = (const float*)d_in[19];
    const float* em_W2    = (const float*)d_in[20];
    const float* em_b2    = (const float*)d_in[21];
    const float* out_W1   = (const float*)d_in[22];
    const float* out_b1   = (const float*)d_in[23];
    const float* out_W2   = (const float*)d_in[24];
    const float* out_b2   = (const float*)d_in[25];
    const float* out_W3   = (const float*)d_in[26];
    const float* out_b3   = (const float*)d_in[27];
    float* out = (float*)d_out;

    // workspace layout (all sizes are 256-byte multiples)
    char* ws = (char*)d_ws;
    size_t off = 0;
    float* ws_irrep  = (float*)(ws + off); off += (size_t)FLAT * IRR * 4;       // 320 KB
    float* ws_rbf    = (float*)(ws + off); off += (size_t)EQ * RBF * 4;         // 8 MB
    float* ws_sh1    = (float*)(ws + off); off += (size_t)EQ * 3 * 4;           // 1.5 MB
    float* ws_envact = (float*)(ws + off); off += (size_t)EQ * 4;               // 0.5 MB
    float* ws_we     = (float*)(ws + off); off += (size_t)EQ * 4;               // 0.5 MB
    float* ws_scale  = (float*)(ws + off); off += (size_t)NEQ * IRR * 4;        // 10 KB
    __bf16* ws_win   = (__bf16*)(ws + off); off += (size_t)EQ * 64 * 2;         // 16 MB
    __bf16* ws_hid   = (__bf16*)(ws + off); off += (size_t)EQ * HID * 2;        // 32 MB

    zero_kernel<<<(FLAT * IRR + 255) / 256, 256, 0, stream>>>(ws_irrep, FLAT * IRR);
    edge_prep_kernel<<<EQ / 256, 256, 0, stream>>>(
        att_dist, att_vec, z, mask, att_src, att_dst, z_emb_W,
        ws_rbf, ws_sh1, ws_envact, ws_win);
    vw_hidden_kernel<<<EQ / 64, 128, 0, stream>>>(ws_win, vw_W1, vw_b1, ws_hid);
    gate_kernel<<<EQ / 64, 128, 0, stream>>>(
        h, gate_W1, gate_b1, gate_W2, gate_b2, ws_rbf, ws_envact,
        att_src, att_dst, ws_we);
    scales_kernel<<<1, 64, 0, stream>>>(e_feat, em_W1, em_b1, em_W2, em_b2, ws_scale);
    edge_contract_kernel<<<EQ / 64, 128, 0, stream>>>(
        ws_hid, vw_W2, vw_b2, h_full, ws_sh1, ws_we, att_src, att_dst, ws_irrep);
    out_mlp_kernel<<<(FLAT * NEQ) / 64, 128, 0, stream>>>(
        ws_irrep, ws_scale, out_W1, out_b1, out_W2, out_b2, out_W3, out_b3, out);
}
